// Decoder_74191265071285
// MI455X (gfx1250) — matrix operand order repaired and checked by reading, not compiled, not run on hardware
//
#include <hip/hip_runtime.h>

// Autoregressive LSTM decode, persistent-tile kernel for gfx1250 (MI455X).
// Grid: 128 blocks x 256 threads (8 wave32). Each block owns 64 batch rows
// for all 64 timesteps.
//   gates  = [x|h] @ [W_ih;W_hh]^T          -> 32 v_wmma_f32_16x16x32_f16 / wave
//   rel    = h' @ W_out^T + b_out           -> 2 WMMAs (waves 0..3, zero-padded N)
//   x_next = h' @ (W_emb@W_out)^T + bfx     -> 4 WMMAs (fused re-embed, exact algebra)
// All weight B-fragments are loop-invariant and live in VGPRs (>256 via the
// gfx1250 VGPR-MSB mechanism). The [x|h] f16 A tile is double-buffered in LDS
// so only 2 workgroup barriers are needed per timestep. Activations use the
// native CDNA5 v_tanh_f32 when the builtin is available.

#define BATCH 8192
#define HID 64
#define EMB 64
#define SEQ 64
#define TB 64          // batch rows per workgroup
#define AXS 136        // A-tile row stride in halfs (272B = 4 banks mod 64 -> conflict free)

typedef __attribute__((ext_vector_type(16))) _Float16 v16h;
typedef __attribute__((ext_vector_type(8)))  _Float16 v8h;
typedef __attribute__((ext_vector_type(8)))  float    v8f;

#if __has_builtin(__builtin_amdgcn_tanhf)
// Native CDNA5 transcendental: 1 instruction per tanh; sigmoid = 3 ops.
__device__ __forceinline__ float tanh_f(float x) {
    return __builtin_amdgcn_tanhf(x);
}
__device__ __forceinline__ float sig_f(float x) {
    return fmaf(0.5f, __builtin_amdgcn_tanhf(0.5f * x), 0.5f);
}
#else
// Fallback: v_exp_f32 + single v_rcp_f32 (no IEEE divide expansion).
__device__ __forceinline__ float sig_f(float x) {
    return __builtin_amdgcn_rcpf(1.0f + __expf(-x));
}
__device__ __forceinline__ float tanh_f(float x) {
    float e = __expf(2.0f * x);           // 1 - 2/(exp(2x)+1): correct limits
    return 1.0f - 2.0f * __builtin_amdgcn_rcpf(e + 1.0f);
}
#endif

__global__ __launch_bounds__(256) void lstm_decode_kernel(
    const float* __restrict__ last_pos, const float* __restrict__ hh,
    const float* __restrict__ ch,       const float* __restrict__ W_emb,
    const float* __restrict__ b_emb,    const float* __restrict__ W_ih,
    const float* __restrict__ W_hh,     const float* __restrict__ b_ih,
    const float* __restrict__ b_hh,     const float* __restrict__ W_out,
    const float* __restrict__ b_out,    float* __restrict__ out)
{
    // ping-pong A tiles: [row][0:64)=x f16, [64:128)=h f16
    __shared__ _Float16 sAxh[2][TB * AXS];

    const int t    = threadIdx.x;
    const int lane = t & 31;
    const int wid  = t >> 5;
    const int g    = lane >> 4;       // half-wave group (WMMA lane split)
    const int col  = lane & 15;       // N column / M row within a 16-tile
    const int mt   = wid & 3;         // M tile (batch rows mt*16..+15)
    const int ntb  = (wid >> 2) << 1; // first of two N tiles for this wave
    const int wgb  = blockIdx.x * TB;
    const int rowoff = (mt * 16 + col) * AXS;   // A-frag source row offset

    // ---- gate-weight B fragments, persistent in VGPRs for all 64 steps ----
    // B is [K=128 x N=256]: B[k][n] = (k<64 ? W_ih[n][k] : W_hh[n][k-64]).
    // Per-lane layout (16-bit B, 32x16): lane holds column n = ntile*16+col,
    // halfs i=0..15 hold K = kc*32 + g*16 + i.
    v16h bfrag[2][4][4];
    #pragma unroll
    for (int p = 0; p < 2; ++p) {
        const int nt = ntb + p;
        #pragma unroll
        for (int q = 0; q < 4; ++q) {
            const int n = q * 64 + nt * 16 + col;
            #pragma unroll
            for (int kc = 0; kc < 4; ++kc) {
                const int k0 = kc * 32 + g * 16;
                const float* src = (k0 < 64) ? (W_ih + (size_t)n * 64 + k0)
                                             : (W_hh + (size_t)n * 64 + (k0 - 64));
                #pragma unroll
                for (int i = 0; i < 16; ++i)
                    bfrag[p][q][kc][i] = (_Float16)src[i];
            }
        }
    }

    // ---- W_out as a zero-padded [K=64 x N=16] B fragment (cols 0,1 valid) ----
    v16h wfrag[2];
    #pragma unroll
    for (int kc = 0; kc < 2; ++kc) {
        const int k0 = kc * 32 + g * 16;
        #pragma unroll
        for (int i = 0; i < 16; ++i)
            wfrag[kc][i] = (col < 2) ? (_Float16)W_out[col * HID + k0 + i]
                                     : (_Float16)0.0f;
    }
    const float boutv = (col < 2) ? b_out[col] : 0.0f;

    // ---- fused re-embed: M = W_emb @ W_out  [K=hid x N=emb], bfx bias ----
    // x_next[b][e] = sum_k h[b][k] * M[k][e] + bfx[e]   (exact refactoring)
    v16h wfx[2][2];
    float xbias[2];
    #pragma unroll
    for (int p = 0; p < 2; ++p) {
        const int n = (ntb + p) * 16 + col;            // embed column
        const float we0 = W_emb[n * 2 + 0];
        const float we1 = W_emb[n * 2 + 1];
        #pragma unroll
        for (int kc = 0; kc < 2; ++kc) {
            const int k0 = kc * 32 + g * 16;
            #pragma unroll
            for (int i = 0; i < 16; ++i)
                wfx[p][kc][i] = (_Float16)(W_out[k0 + i] * we0 +
                                           W_out[HID + k0 + i] * we1);
        }
        xbias[p] = fmaf(b_out[0], we0, fmaf(b_out[1], we1, b_emb[n]));
    }

    // combined gate biases, one per (position, gate) per lane
    float bias[2][4];
    #pragma unroll
    for (int p = 0; p < 2; ++p)
        #pragma unroll
        for (int q = 0; q < 4; ++q) {
            const int n = q * 64 + (ntb + p) * 16 + col;
            bias[p][q] = b_ih[n] + b_hh[n];
        }

    // cell state c + last hidden h in registers, WMMA C/D layout:
    // VGPR r <-> batch row mt*16 + g*8 + r, hid = nt*16 + col
    float creg[2][8], hreg[2][8];
    #pragma unroll
    for (int p = 0; p < 2; ++p)
        #pragma unroll
        for (int r = 0; r < 8; ++r)
            creg[p][r] = ch[(size_t)(wgb + mt * 16 + g * 8 + r) * HID +
                            (ntb + p) * 16 + col];

    // h0 (f16 into buffer 0) and x0 = last_pos_rel @ W_emb^T + b_emb
    {
        const int row = t >> 2;
        const int c0  = (t & 3) * 16;
        #pragma unroll
        for (int i = 0; i < 16; ++i) {
            float v = hh[(size_t)(wgb + row) * HID + c0 + i];
            sAxh[0][row * AXS + 64 + c0 + i] = (_Float16)v;
        }
        const float r0 = last_pos[(size_t)(wgb + row) * 2 + 0];
        const float r1 = last_pos[(size_t)(wgb + row) * 2 + 1];
        #pragma unroll
        for (int i = 0; i < 16; ++i) {
            const int e = c0 + i;
            float xv = fmaf(r0, W_emb[e * 2 + 0],
                       fmaf(r1, W_emb[e * 2 + 1], b_emb[e]));
            sAxh[0][row * AXS + e] = (_Float16)xv;
        }
    }
    __syncthreads();

    for (int s = 0; s < SEQ; ++s) {
        const int rd = s & 1;
        const int wr = rd ^ 1;
        const _Float16* aR = &sAxh[rd][rowoff];

        // ---- gate A fragments: lane group g holds K = kc*32 + g*16 + 0..15 ----
        v16h af[4];
        #pragma unroll
        for (int kc = 0; kc < 4; ++kc) {
            const int k0 = kc * 32 + g * 16;
            v8h lo = *(const v8h*)(aR + k0);          // 16B aligned ds_load_b128
            v8h hi = *(const v8h*)(aR + k0 + 8);
            #pragma unroll
            for (int i = 0; i < 8; ++i) { af[kc][i] = lo[i]; af[kc][i + 8] = hi[i]; }
        }

        // ---- gates = [x|h] @ Wcat, accumulators pre-seeded with bias ----
        v8f acc[2][4];
        #pragma unroll
        for (int p = 0; p < 2; ++p)
            #pragma unroll
            for (int q = 0; q < 4; ++q)
                #pragma unroll
                for (int i = 0; i < 8; ++i) acc[p][q][i] = bias[p][q];

        #pragma unroll
        for (int kc = 0; kc < 4; ++kc)
            #pragma unroll
            for (int p = 0; p < 2; ++p)
                #pragma unroll
                for (int q = 0; q < 4; ++q)
                    acc[p][q] = __builtin_amdgcn_wmma_f32_16x16x32_f16(
                        false, af[kc], false, bfrag[p][q][kc],
                        (short)0, acc[p][q], false, false);

        // ---- LSTM cell (i,f,g,o), update c; h' -> write buffer (no barrier:
        //      reads were from the read buffer, writes go to the other one) ----
        #pragma unroll
        for (int p = 0; p < 2; ++p) {
            const int nt = ntb + p;
            #pragma unroll
            for (int r = 0; r < 8; ++r) {
                float iv = sig_f (acc[p][0][r]);
                float fv = sig_f (acc[p][1][r]);
                float gv = tanh_f(acc[p][2][r]);
                float ov = sig_f (acc[p][3][r]);
                float cv = fmaf(fv, creg[p][r], iv * gv);
                creg[p][r] = cv;
                float hv = ov * tanh_f(cv);
                hreg[p][r] = hv;
                const int row = mt * 16 + g * 8 + r;
                sAxh[wr][row * AXS + 64 + nt * 16 + col] = (_Float16)hv;
            }
        }
        __syncthreads();   // h' visible to all waves

        // ---- shared h' A-fragments (K = 64) from the write buffer ----
        {
            const _Float16* hR = &sAxh[wr][rowoff] + 64;
            v16h hf[2];
            #pragma unroll
            for (int kc = 0; kc < 2; ++kc) {
                const int k0 = kc * 32 + g * 16;
                v8h lo = *(const v8h*)(hR + k0);
                v8h hi = *(const v8h*)(hR + k0 + 8);
                #pragma unroll
                for (int i = 0; i < 8; ++i) { hf[kc][i] = lo[i]; hf[kc][i + 8] = hi[i]; }
            }

            // x_next = h' @ M + bfx  (fused re-embed), back into write buffer
            #pragma unroll
            for (int p = 0; p < 2; ++p) {
                v8f xacc;
                #pragma unroll
                for (int i = 0; i < 8; ++i) xacc[i] = xbias[p];
                #pragma unroll
                for (int kc = 0; kc < 2; ++kc)
                    xacc = __builtin_amdgcn_wmma_f32_16x16x32_f16(
                        false, hf[kc], false, wfx[p][kc], (short)0, xacc, false, false);
                #pragma unroll
                for (int r = 0; r < 8; ++r)
                    sAxh[wr][(mt * 16 + g * 8 + r) * AXS + (ntb + p) * 16 + col] =
                        (_Float16)xacc[r];
            }

            // rel_pos = h' @ W_out^T + b_out  (waves 0..3), trajectory output
            if (wid < 4) {
                v8f racc;
                #pragma unroll
                for (int i = 0; i < 8; ++i) racc[i] = boutv;
                #pragma unroll
                for (int kc = 0; kc < 2; ++kc)
                    racc = __builtin_amdgcn_wmma_f32_16x16x32_f16(
                        false, hf[kc], false, wfrag[kc], (short)0, racc, false, false);
                if (col < 2) {
                    #pragma unroll
                    for (int r = 0; r < 8; ++r) {
                        const int row = mt * 16 + g * 8 + r;
                        out[((size_t)s * BATCH + wgb + row) * 2 + col] = racc[r];
                    }
                }
            }
        }
        __syncthreads();   // x' visible before next step's gate GEMM
    }

    // ---- final hidden state [1, B, H], straight from registers ----
    {
        float* outH = out + (size_t)SEQ * BATCH * 2;
        #pragma unroll
        for (int p = 0; p < 2; ++p)
            #pragma unroll
            for (int r = 0; r < 8; ++r)
                outH[(size_t)(wgb + mt * 16 + g * 8 + r) * HID +
                     (ntb + p) * 16 + col] = hreg[p][r];
    }
}

extern "C" void kernel_launch(void* const* d_in, const int* in_sizes, int n_in,
                              void* d_out, int out_size, void* d_ws, size_t ws_size,
                              hipStream_t stream) {
    (void)in_sizes; (void)n_in; (void)out_size; (void)d_ws; (void)ws_size;
    lstm_decode_kernel<<<BATCH / TB, 256, 0, stream>>>(
        (const float*)d_in[0],  // last_pos_rel [B,2]
        (const float*)d_in[1],  // hh [1,B,H]
        (const float*)d_in[2],  // ch [1,B,H]
        (const float*)d_in[3],  // W_emb [E,2]
        (const float*)d_in[4],  // b_emb [E]
        (const float*)d_in[5],  // W_ih [4H,E]
        (const float*)d_in[6],  // W_hh [4H,H]
        (const float*)d_in[7],  // b_ih [4H]
        (const float*)d_in[8],  // b_hh [4H]
        (const float*)d_in[9],  // W_out [2,H]
        (const float*)d_in[10], // b_out [2]
        (float*)d_out);
}
